// SparseConvBlock_5549097746516
// MI455X (gfx1250) — matrix operand order repaired and checked by reading, not compiled, not run on hardware
//
#include <hip/hip_runtime.h>
#include <hip/hip_bf16.h>
#include <hip/hip_fp16.h>

// ---------------------------------------------------------------------------
// Problem constants (match reference)
// ---------------------------------------------------------------------------
#define N_IN    262144
#define N_OUT   262144
#define KOFF    27
#define MPAIR   131072
#define C_IN    64
#define C_OUT   64
#define EPS     1e-5f
#define SLOPE   0.01f

typedef __attribute__((ext_vector_type(8)))  _Float16 v8h;
typedef __attribute__((ext_vector_type(16))) _Float16 v16h;
typedef __attribute__((ext_vector_type(8)))  float    v8f;

// hardware fp32 atomic add, no return (global_atomic_add_f32)
__device__ __forceinline__ void atomAddF32(float* p, float v) {
    unsafeAtomicAdd(p, v);
}

// ---------------------------------------------------------------------------
// 1) zero the f32 accumulator (d_out doubles as accumulator)
// ---------------------------------------------------------------------------
__global__ void k_zero_f32(float4* p, int n4) {
    int i = blockIdx.x * blockDim.x + threadIdx.x;
    if (i < n4) p[i] = make_float4(0.f, 0.f, 0.f, 0.f);
}

// ---------------------------------------------------------------------------
// 2) feats f32 -> f16 (row-major copy, halves gather traffic in hot loop)
// ---------------------------------------------------------------------------
__global__ void k_feats_to_h(const float4* __restrict__ f, _Float16* __restrict__ fh, int n4) {
    int i = blockIdx.x * blockDim.x + threadIdx.x;
    if (i >= n4) return;
    float4 v = f[i];
    _Float16* o = fh + (size_t)i * 4;
    o[0] = (_Float16)v.x; o[1] = (_Float16)v.y; o[2] = (_Float16)v.z; o[3] = (_Float16)v.w;
}

// ---------------------------------------------------------------------------
// 3) W f32 -> f16 pre-swizzled to WMMA B-operand layout + zero BN stats.
//    wb[k][c2][n][lane][e]  (e=0..15 contiguous per lane)
//    value = W[k][ c2*32 + (lane<16?0:8) + (e<8?e:e+8) ][ n*16 + (lane&15) ]
// ---------------------------------------------------------------------------
__global__ void k_w_swizzle(const float* __restrict__ W, _Float16* __restrict__ wb,
                            float* __restrict__ stats) {
    int id = blockIdx.x * blockDim.x + threadIdx.x;
    if (id < 128) stats[id] = 0.f;                    // gsum[64], gsq[64]
    if (id >= KOFF * 2 * 4 * 32 * 16) return;         // 110592 elements
    int e    = id & 15;
    int lane = (id >> 4) & 31;
    int n    = (id >> 9) & 3;
    int c2   = (id >> 11) & 1;
    int k    = id >> 12;
    int col  = n * 16 + (lane & 15);
    int krow = c2 * 32 + ((lane < 16) ? 0 : 8) + ((e < 8) ? e : e + 8);
    wb[id] = (_Float16)W[(size_t)k * (C_IN * C_OUT) + (size_t)krow * C_OUT + col];
}

// ---------------------------------------------------------------------------
// 4) Main sparse-conv: gather -> v_wmma_f32_16x16x32_f16 -> atomic scatter
//    grid = (64, 27), block = 256 (8 waves). Each wave: 16 tiles of 16 pairs.
// ---------------------------------------------------------------------------
#define CONV_BLKX 64
#define WAVES_PER_BLK 8
#define TILES_PER_K (MPAIR / 16)   // 8192

__global__ void __launch_bounds__(256)
k_spconv(const _Float16* __restrict__ fh,
         const _Float16* __restrict__ wb,
         const int* __restrict__ in_map,
         const int* __restrict__ out_map,
         float* __restrict__ acc) {
    const int k    = blockIdx.y;
    const int wave = threadIdx.x >> 5;
    const int lid  = threadIdx.x & 31;
    const int l16  = lid & 15;
    const int hi   = (lid < 16) ? 0 : 1;

    // ---- load the 8 B fragments for this kernel offset once (64 VGPRs) ----
    const _Float16* wbk = wb + (size_t)k * (8 * 32 * 16);
    v16h B[2][4];
#pragma unroll
    for (int c2 = 0; c2 < 2; ++c2)
#pragma unroll
        for (int n = 0; n < 4; ++n)
            B[c2][n] = *(const v16h*)(wbk + ((size_t)((c2 * 4 + n) * 32 + lid)) * 16);

    const int* imk = in_map  + (size_t)k * MPAIR;
    const int* omk = out_map + (size_t)k * MPAIR;

    for (int t = blockIdx.x * WAVES_PER_BLK + wave; t < TILES_PER_K;
         t += CONV_BLKX * WAVES_PER_BLK) {
        const int m0 = t * 16;

        // ---- gather A fragments (ISA 16-bit 16x32 A layout) ----
        const int im = imk[m0 + l16];
        const _Float16* row = fh + (size_t)im * C_IN;
        const int off = hi ? 8 : 0;
        v8h a0l = *(const v8h*)(row +  0 + off);
        v8h a0h = *(const v8h*)(row + 16 + off);
        v8h a1l = *(const v8h*)(row + 32 + off);
        v8h a1h = *(const v8h*)(row + 48 + off);
        v16h A0 = __builtin_shufflevector(a0l, a0h, 0,1,2,3,4,5,6,7,8,9,10,11,12,13,14,15);
        v16h A1 = __builtin_shufflevector(a1l, a1h, 0,1,2,3,4,5,6,7,8,9,10,11,12,13,14,15);

        // ---- 8x v_wmma_f32_16x16x32_f16 : 16 pairs x 64 Cin x 64 Cout ----
        v8f C0 = {}, C1 = {}, C2 = {}, C3 = {};
        C0 = __builtin_amdgcn_wmma_f32_16x16x32_f16(false, A0, false, B[0][0], (short)0, C0, false, false);
        C1 = __builtin_amdgcn_wmma_f32_16x16x32_f16(false, A0, false, B[0][1], (short)0, C1, false, false);
        C2 = __builtin_amdgcn_wmma_f32_16x16x32_f16(false, A0, false, B[0][2], (short)0, C2, false, false);
        C3 = __builtin_amdgcn_wmma_f32_16x16x32_f16(false, A0, false, B[0][3], (short)0, C3, false, false);
        C0 = __builtin_amdgcn_wmma_f32_16x16x32_f16(false, A1, false, B[1][0], (short)0, C0, false, false);
        C1 = __builtin_amdgcn_wmma_f32_16x16x32_f16(false, A1, false, B[1][1], (short)0, C1, false, false);
        C2 = __builtin_amdgcn_wmma_f32_16x16x32_f16(false, A1, false, B[1][2], (short)0, C2, false, false);
        C3 = __builtin_amdgcn_wmma_f32_16x16x32_f16(false, A1, false, B[1][3], (short)0, C3, false, false);

        // ---- scatter with hw f32 atomics; C layout: elem v -> pair v + 8*hi-half
        const int om = omk[m0 + l16];
#pragma unroll
        for (int v = 0; v < 8; ++v) {
            int srcLane = v + (hi ? 8 : 0);
            int orow = __shfl(om, srcLane, 32);
            float* base = acc + (size_t)orow * C_OUT + l16;
            atomAddF32(base +  0, C0[v]);
            atomAddF32(base + 16, C1[v]);
            atomAddF32(base + 32, C2[v]);
            atomAddF32(base + 48, C3[v]);
        }
    }
}

// ---------------------------------------------------------------------------
// 5) BN statistics: per-channel sum and sum-of-squares over N_OUT rows
// ---------------------------------------------------------------------------
__global__ void __launch_bounds__(256)
k_bn_stats(const float* __restrict__ acc, float* __restrict__ gsum, float* __restrict__ gsq) {
    __shared__ float ssum[256];
    __shared__ float ssq[256];
    const int tid = threadIdx.x;
    const int c   = tid & 63;
    const int rg  = tid >> 6;                 // 4 row groups per block
    float s = 0.f, q = 0.f;
    for (int r = blockIdx.x * 4 + rg; r < N_OUT; r += gridDim.x * 4) {
        float v = acc[(size_t)r * C_OUT + c];
        s += v; q += v * v;
    }
    ssum[tid] = s; ssq[tid] = q;
    __syncthreads();
    if (tid < 64) {
        float S = ssum[tid] + ssum[tid + 64] + ssum[tid + 128] + ssum[tid + 192];
        float Q = ssq[tid]  + ssq[tid + 64]  + ssq[tid + 128]  + ssq[tid + 192];
        atomAddF32(&gsum[tid], S);
        atomAddF32(&gsq[tid],  Q);
    }
}

// ---------------------------------------------------------------------------
// 6) fold BN into per-channel scale/bias
// ---------------------------------------------------------------------------
__global__ void k_bn_finalize(const float* __restrict__ gsum, const float* __restrict__ gsq,
                              const float* __restrict__ gamma, const float* __restrict__ beta,
                              float* __restrict__ scale, float* __restrict__ bias) {
    int c = threadIdx.x;
    if (c >= C_OUT) return;
    const float invN = 1.0f / (float)N_OUT;
    float mean = gsum[c] * invN;
    float var  = gsq[c] * invN - mean * mean;
    float sc   = gamma[c] * rsqrtf(var + EPS);
    scale[c] = sc;
    bias[c]  = beta[c] - mean * sc;
}

// ---------------------------------------------------------------------------
// 7) apply BN + LeakyReLU in place (float4 vectorized)
// ---------------------------------------------------------------------------
__global__ void __launch_bounds__(256)
k_bn_apply(float* __restrict__ out, const float* __restrict__ scale,
           const float* __restrict__ bias, int n4) {
    int i = blockIdx.x * blockDim.x + threadIdx.x;
    if (i >= n4) return;
    float4 v = ((const float4*)out)[i];
    int c0 = (i * 4) & 63;
    float y0 = v.x * scale[c0 + 0] + bias[c0 + 0];
    float y1 = v.y * scale[c0 + 1] + bias[c0 + 1];
    float y2 = v.z * scale[c0 + 2] + bias[c0 + 2];
    float y3 = v.w * scale[c0 + 3] + bias[c0 + 3];
    v.x = (y0 >= 0.f) ? y0 : SLOPE * y0;
    v.y = (y1 >= 0.f) ? y1 : SLOPE * y1;
    v.z = (y2 >= 0.f) ? y2 : SLOPE * y2;
    v.w = (y3 >= 0.f) ? y3 : SLOPE * y3;
    ((float4*)out)[i] = v;
}

// ---------------------------------------------------------------------------
// launcher
// ---------------------------------------------------------------------------
extern "C" void kernel_launch(void* const* d_in, const int* in_sizes, int n_in,
                              void* d_out, int out_size, void* d_ws, size_t ws_size,
                              hipStream_t stream) {
    const float* feats   = (const float*)d_in[0];
    const float* W       = (const float*)d_in[1];
    const float* gamma   = (const float*)d_in[2];
    const float* beta    = (const float*)d_in[3];
    const int*   in_map  = (const int*)d_in[4];
    const int*   out_map = (const int*)d_in[5];
    float* acc = (float*)d_out;                     // f32 accumulator == output

    // workspace layout
    char* ws = (char*)d_ws;
    _Float16* featsH = (_Float16*)ws;                               // 32 MB
    size_t off = (size_t)N_IN * C_IN * sizeof(_Float16);
    _Float16* wb = (_Float16*)(ws + off);                           // 216 KB
    off += (size_t)KOFF * 2 * 4 * 32 * 16 * sizeof(_Float16);
    float* gsum  = (float*)(ws + off);           // 64
    float* gsq   = gsum + 64;                    // 64
    float* scale = gsum + 128;                   // 64
    float* bias  = gsum + 192;                   // 64

    const int nElems = N_OUT * C_OUT;            // 16,777,216
    const int n4     = nElems / 4;               // 4,194,304

    // 1) zero accumulator
    k_zero_f32<<<(n4 + 255) / 256, 256, 0, stream>>>((float4*)acc, n4);
    // 2) feats -> f16
    k_feats_to_h<<<((N_IN * C_IN / 4) + 255) / 256, 256, 0, stream>>>(
        (const float4*)feats, featsH, N_IN * C_IN / 4);
    // 3) W -> f16 B-layout, zero stats
    k_w_swizzle<<<(KOFF * 2 * 4 * 32 * 16 + 255) / 256, 256, 0, stream>>>(W, wb, gsum);
    // 4) sparse conv (gather - WMMA - atomic scatter)
    dim3 cgrid(CONV_BLKX, KOFF);
    k_spconv<<<cgrid, 256, 0, stream>>>(featsH, wb, in_map, out_map, acc);
    // 5) BN stats
    k_bn_stats<<<1024, 256, 0, stream>>>(acc, gsum, gsq);
    // 6) finalize scale/bias
    k_bn_finalize<<<1, 64, 0, stream>>>(gsum, gsq, gamma, beta, scale, bias);
    // 7) apply BN + LeakyReLU in place
    k_bn_apply<<<(n4 + 255) / 256, 256, 0, stream>>>(acc, scale, bias, n4);
}
